// LSTMC_83915071030074
// MI455X (gfx1250) — compile-verified, matrix operand order checked by reading, not run on hardware
//
#include <hip/hip_runtime.h>
#include <hip/hip_bf16.h>

typedef __attribute__((ext_vector_type(16))) __bf16 v16bf;
typedef __attribute__((ext_vector_type(8)))  float  v8f;

#define T_   2048
#define B_   256
#define E_   128
#define H_   256
#define G_   1024
#define NB   64
#define FRAG 512          // elements per 16x16x32 fragment (32 lanes * 16)
#define KT_X 4            // E/32 k-fragments
#define KT_H 8            // H/32 k-fragments
#define NT   64           // G/16 n-tiles
#define MT   16           // B/16 m-tiles
#define HB   (MT*KT_H*FRAG)   // 65536 bf16 elems per h buffer

// workspace byte offsets
#define OFF_BAR 0
#define OFF_H   256                              // 2*HB bf16     = 262144 B
#define OFF_WIH (OFF_H + 2*HB*2)                 // NT*KT_X*FRAG  = 262144 B
#define OFF_WHH (OFF_WIH + NT*KT_X*FRAG*2)       // NT*KT_H*FRAG  = 524288 B
#define OFF_X   (OFF_WHH + NT*KT_H*FRAG*2)       // T*MT*KT_X*FRAG bf16 = 128 MB

__device__ __forceinline__ unsigned short f2bf(float f) {
    unsigned int u = __float_as_uint(f);
    unsigned int r = (u + 0x7FFFu + ((u >> 16) & 1u)) >> 16;   // RNE
    return (unsigned short)r;
}
__device__ __forceinline__ float bf2f(unsigned short h) {
    return __uint_as_float(((unsigned int)h) << 16);
}
// A/B 16-bit fragment K mapping: element e (0..15) of lane-half h -> local k
__device__ __forceinline__ int kmap(int half, int e) {
    return (e < 8) ? (e + 8 * half) : (16 + (e - 8) + 8 * half);
}
__device__ __forceinline__ float fsig(float x) { return 1.0f / (1.0f + __expf(-x)); }
__device__ __forceinline__ float ftanh(float x) {
    float e = __expf(2.0f * x);
    return (e - 1.0f) / (e + 1.0f);
}

// ---- pack W_ih / W_hh into pre-swizzled bf16 B-fragments ----
__global__ void lstm_pack(const float* __restrict__ W_ih, const float* __restrict__ W_hh,
                          unsigned short* __restrict__ wp_ih, unsigned short* __restrict__ wp_hh) {
    int idx = blockIdx.x * blockDim.x + threadIdx.x;
    const int NIH = NT * KT_X * FRAG;   // 131072
    const int NHH = NT * KT_H * FRAG;   // 262144
    if (idx < NIH) {
        int e = idx & 15, lane = (idx >> 4) & 31, kt = (idx >> 9) & (KT_X - 1), nt = idx >> 11;
        int n = nt * 16 + (lane & 15);
        int k = kt * 32 + kmap(lane >> 4, e);
        wp_ih[idx] = f2bf(W_ih[n * E_ + k]);
    } else if (idx < NIH + NHH) {
        int j = idx - NIH;
        int e = j & 15, lane = (j >> 4) & 31, kt = (j >> 9) & (KT_H - 1), nt = j >> 12;
        int n = nt * 16 + (lane & 15);
        int k = kt * 32 + kmap(lane >> 4, e);
        wp_hh[j] = f2bf(W_hh[n * H_ + k]);
    }
}

// ---- embedding gather -> bf16 x in A-fragment swizzled layout ----
__global__ void lstm_gather(const int* __restrict__ inputs, const float* __restrict__ emb,
                            unsigned short* __restrict__ xs) {
    int t = blockIdx.x;
    int b = threadIdx.x;
    int tok = inputs[t * B_ + b];
    const float* er = emb + (size_t)tok * E_;
    int mm = b & 15, mt = b >> 4;
    unsigned short* base = xs + (size_t)(t * MT + mt) * (KT_X * FRAG);
    for (int k = 0; k < E_; ++k) {
        int kt = k >> 5, ke = k & 31;
        int half = (ke & 16) ? ((ke - 16) >> 3) : (ke >> 3);
        int e    = (ke & 16) ? (8 + (ke & 7)) : (ke & 7);
        int lane = mm + 16 * half;
        base[kt * FRAG + lane * 16 + e] = f2bf(er[k]);
    }
}

// ---- persistent recurrent kernel: 2048 steps, grid barrier per step ----
__global__ void __launch_bounds__(128, 1) __attribute__((amdgpu_num_vgpr(1024)))
lstm_persistent(const unsigned short* __restrict__ xs,
                const unsigned short* __restrict__ wih,
                const unsigned short* __restrict__ whh,
                const float* __restrict__ b_ih, const float* __restrict__ b_hh,
                unsigned short* __restrict__ hbuf,
                unsigned int* __restrict__ bar) {
    const int nb   = blockIdx.x;
    const int gidx = nb & 15;                               // h-column group (16 cols)
    const int mt   = ((nb >> 4) << 2) + (threadIdx.x >> 5); // this wave's m-tile
    const int lane = threadIdx.x & 31;
    const int ncol = lane & 15;
    const int half = lane >> 4;
    const int j    = gidx * 16 + ncol;                      // h column owned by this lane

    float bsum[4];
#pragma unroll
    for (int g = 0; g < 4; ++g) bsum[g] = b_ih[g * H_ + j] + b_hh[g * H_ + j];

    // h-store swizzle for column j (k = j in the h A-layout)
    const int hkt   = j >> 5;
    const int hke   = j & 31;
    const int hhalf = (hke & 16) ? ((hke - 16) >> 3) : (hke >> 3);
    const int he    = (hke & 16) ? (8 + (hke & 7)) : (hke & 7);

    const unsigned short* wi[4];
    const unsigned short* wh[4];
#pragma unroll
    for (int g = 0; g < 4; ++g) {
        int nt = g * 16 + gidx;
        wi[g] = wih + (size_t)nt * (KT_X * FRAG) + lane * 16;
        wh[g] = whh + (size_t)nt * (KT_H * FRAG) + lane * 16;
    }

    // cell state lives entirely in registers: this thread owns (m = mt*16+v+8*half, j)
    float creg[8];
#pragma unroll
    for (int v = 0; v < 8; ++v) creg[v] = 0.0f;

    // prefetch x fragments for t = 0
    v16bf ax[KT_X];
    {
        const unsigned short* xf = xs + (size_t)(0 * MT + mt) * (KT_X * FRAG) + lane * 16;
#pragma unroll
        for (int kt = 0; kt < KT_X; ++kt)
            ax[kt] = *reinterpret_cast<const v16bf*>(xf + kt * FRAG);
    }

    for (int t = 0; t < T_; ++t) {
        const unsigned short* hin  = hbuf + (size_t)(t & 1) * HB;
        unsigned short*       hout = hbuf + (size_t)((t + 1) & 1) * HB;
        const unsigned short* hf   = hin + (size_t)mt * (KT_H * FRAG) + lane * 16;

        // issue ALL h-fragment loads as one clause; pin them here so the
        // scheduler cannot sink them to their use sites
        v16bf ah[KT_H];
#pragma unroll
        for (int kt = 0; kt < KT_H; ++kt)
            ah[kt] = *reinterpret_cast<const v16bf*>(hf + kt * FRAG);
        __builtin_amdgcn_sched_barrier(0);

        v8f acc[4];
#pragma unroll
        for (int g = 0; g < 4; ++g)
#pragma unroll
            for (int v = 0; v < 8; ++v) acc[g][v] = bsum[g];

        // input-side: g += x_t · W_ih^T (x frags already resident; h loads in flight)
#pragma unroll
        for (int kt = 0; kt < KT_X; ++kt)
#pragma unroll
            for (int g = 0; g < 4; ++g)
                acc[g] = __builtin_amdgcn_wmma_f32_16x16x32_bf16(
                    false, ax[kt], false, *reinterpret_cast<const v16bf*>(wi[g] + kt * FRAG),
                    (short)0, acc[g], false, false);

        // recurrent side: g += h · W_hh^T
#pragma unroll
        for (int kt = 0; kt < KT_H; ++kt)
#pragma unroll
            for (int g = 0; g < 4; ++g)
                acc[g] = __builtin_amdgcn_wmma_f32_16x16x32_bf16(
                    false, ah[kt], false, *reinterpret_cast<const v16bf*>(wh[g] + kt * FRAG),
                    (short)0, acc[g], false, false);

        // gate math: lane holds i/f/g/o for the SAME (m, j) -> no shuffles needed
        unsigned short* hob = hout + (size_t)(mt * KT_H + hkt) * FRAG + (16 * hhalf) * 16 + he;
#pragma unroll
        for (int v = 0; v < 8; ++v) {
            int m = mt * 16 + v + 8 * half;  // batch row
            float iv = fsig(acc[0][v]);
            float fv = fsig(acc[1][v]);
            float gv = ftanh(acc[2][v]);
            float ov = fsig(acc[3][v]);
            creg[v] = fv * creg[v] + iv * gv;
            hob[(m & 15) * 16] = f2bf(ov * ftanh(creg[v]));
        }

        // prefetch next step's x fragments (x is read-only: safe before the barrier,
        // latency hidden behind the barrier spin)
        int tn = (t + 1 < T_) ? (t + 1) : t;
        const unsigned short* xfn = xs + (size_t)(tn * MT + mt) * (KT_X * FRAG) + lane * 16;
#pragma unroll
        for (int kt = 0; kt < KT_X; ++kt)
            ax[kt] = *reinterpret_cast<const v16bf*>(xfn + kt * FRAG);
        __builtin_amdgcn_sched_barrier(0);

        // device-wide step barrier (monotonic counter, release/acquire)
        __syncthreads();
        if (threadIdx.x == 0) {
            __hip_atomic_fetch_add(bar, 1u, __ATOMIC_RELEASE, __HIP_MEMORY_SCOPE_AGENT);
            unsigned int target = (unsigned int)(t + 1) * NB;
            while (__hip_atomic_load(bar, __ATOMIC_ACQUIRE, __HIP_MEMORY_SCOPE_AGENT) < target)
                __builtin_amdgcn_s_sleep(1);
        }
        __syncthreads();
    }
}

// ---- final: out[b] = sigmoid(hT[b,:] . W_out + b_out) ----
__global__ void lstm_out(const unsigned short* __restrict__ hT,  // buffer 0 (t=2047 wrote buf (2048&1)=0)
                         const float* __restrict__ W_out, const float* __restrict__ b_out,
                         float* __restrict__ out) {
    int b = threadIdx.x;
    int mm = b & 15, mt = b >> 4;
    float s = b_out[0];
    for (int j = 0; j < H_; ++j) {
        int kt = j >> 5, ke = j & 31;
        int half = (ke & 16) ? ((ke - 16) >> 3) : (ke >> 3);
        int e    = (ke & 16) ? (8 + (ke & 7)) : (ke & 7);
        int lane = mm + 16 * half;
        s += bf2f(hT[(size_t)(mt * KT_H + kt) * FRAG + lane * 16 + e]) * W_out[j];
    }
    out[b] = 1.0f / (1.0f + __expf(-s));
}

extern "C" void kernel_launch(void* const* d_in, const int* in_sizes, int n_in,
                              void* d_out, int out_size, void* d_ws, size_t ws_size,
                              hipStream_t stream) {
    const int*   ids   = (const int*)d_in[0];
    const float* emb   = (const float*)d_in[1];
    const float* W_ih  = (const float*)d_in[2];
    const float* W_hh  = (const float*)d_in[3];
    const float* b_ih  = (const float*)d_in[4];
    const float* b_hh  = (const float*)d_in[5];
    const float* W_out = (const float*)d_in[6];
    const float* b_out = (const float*)d_in[7];

    char* ws = (char*)d_ws;
    unsigned int*   bar  = (unsigned int*)(ws + OFF_BAR);
    unsigned short* hbuf = (unsigned short*)(ws + OFF_H);
    unsigned short* wih  = (unsigned short*)(ws + OFF_WIH);
    unsigned short* whh  = (unsigned short*)(ws + OFF_WHH);
    unsigned short* xs   = (unsigned short*)(ws + OFF_X);

    // zero barrier + h0 each call (deterministic across graph replays)
    hipMemsetAsync(d_ws, 0, OFF_WIH, stream);

    lstm_pack<<<1536, 256, 0, stream>>>(W_ih, W_hh, wih, whh);
    lstm_gather<<<T_, B_, 0, stream>>>(ids, emb, xs);
    lstm_persistent<<<NB, 128, 0, stream>>>(xs, wih, whh, b_ih, b_hh, hbuf, bar);
    lstm_out<<<1, B_, 0, stream>>>(hbuf /* buffer 0 holds hT */, W_out, b_out, (float*)d_out);
}
